// MultiheadedAttentionNode_64201171141376
// MI455X (gfx1250) — compile-verified, matrix operand order checked by reading, not run on hardware
//
#include <hip/hip_runtime.h>
#include <hip/hip_bf16.h>
#include <math.h>

#define B_   4
#define S_   2048
#define D_   1024
#define H_   16
#define HD_  64
#define MTOT (B_ * S_)   // 8192

typedef __attribute__((ext_vector_type(16))) __bf16 v16bf;
typedef __attribute__((ext_vector_type(8)))  float  v8f;

union FragB {
    v16bf v;
    uint4 q[2];
    unsigned short s[16];
};

__device__ __forceinline__ unsigned short f2bf(float f) {
    union { float f; unsigned int u; } x; x.f = f;
    unsigned int u = x.u;
    return (unsigned short)((u + 0x7FFFu + ((u >> 16) & 1u)) >> 16);
}

// ---------------------------------------------------------------------------
// Elementwise f32 -> bf16, 8 elements/thread, packed b128 stores.
// ---------------------------------------------------------------------------
__global__ void cvt_f32_bf16_vec8(const float* __restrict__ X,
                                  unsigned short* __restrict__ Y, int n8) {
    int idx = blockIdx.x * blockDim.x + threadIdx.x;
    if (idx >= n8) return;
    const float* p = X + (size_t)idx * 8;
    unsigned int w[4];
    #pragma unroll
    for (int i = 0; i < 4; i++) {
        unsigned int lo = f2bf(p[2 * i]);
        unsigned int hi = f2bf(p[2 * i + 1]);
        w[i] = lo | (hi << 16);
    }
    uint4 o; o.x = w[0]; o.y = w[1]; o.z = w[2]; o.w = w[3];
    *(uint4*)(Y + (size_t)idx * 8) = o;
}

// ---------------------------------------------------------------------------
// Weight transpose + f32->bf16: WT[n*K+k] = bf16(W[k*N+n])
// ---------------------------------------------------------------------------
__global__ void wt_transpose_bf16(const float* __restrict__ W,
                                  unsigned short* __restrict__ WT,
                                  int N, int K) {
    int idx = blockIdx.x * blockDim.x + threadIdx.x;
    if (idx >= N * K) return;
    int n = idx / K, k = idx % K;
    WT[idx] = f2bf(W[(size_t)k * N + n]);
}

// ---------------------------------------------------------------------------
// WMMA GEMM: C[M,N] = A[M,K] @ B[K,N] + bias; A bf16 [M][K], B bf16 transposed
// [N][K]. Wave tile 32x64: 2 A-frags + 4 B-frags -> 8 WMMAs per k-step.
// Block = 8 waves (4 in M x 2 in N) = 128x128 tile.
// OUT_MODE: 0 = bf16 natural [M][N]; 1 = bf16 V-transposed [B][H][HD][S];
//           2 = fp32 natural [M][N].
// ---------------------------------------------------------------------------
template <int OUT_MODE>
__global__ __launch_bounds__(256) void gemm32x64(
    const unsigned short* __restrict__ Abf, const unsigned short* __restrict__ BT,
    const float* __restrict__ bias, void* __restrict__ Cout,
    int Mdim, int Ndim, int Kdim)
{
    const int lane = threadIdx.x & 31;
    const int wave = threadIdx.x >> 5;
    const int half = lane >> 4;    // which K-half of the fragment
    const int l16  = lane & 15;
    const int wm   = wave & 3;     // 4 waves tile M
    const int wn   = wave >> 2;    // 2 waves tile N
    const int m0 = blockIdx.y * 128 + wm * 32;
    const int n0 = blockIdx.x * 128 + wn * 64;

    v8f acc[2][4];
    #pragma unroll
    for (int mt = 0; mt < 2; mt++)
        #pragma unroll
        for (int nc = 0; nc < 4; nc++) acc[mt][nc] = {};

    const unsigned short* arow0 = Abf + (size_t)(m0 + l16)      * Kdim + half * 8;
    const unsigned short* arow1 = Abf + (size_t)(m0 + 16 + l16) * Kdim + half * 8;

    for (int k = 0; k < Kdim; k += 32) {
        FragB a0, a1;
        a0.q[0] = *(const uint4*)(arow0 + k);
        a0.q[1] = *(const uint4*)(arow0 + k + 16);
        a1.q[0] = *(const uint4*)(arow1 + k);
        a1.q[1] = *(const uint4*)(arow1 + k + 16);
        #pragma unroll
        for (int nc = 0; nc < 4; nc++) {
            const unsigned short* bp =
                BT + (size_t)(n0 + nc * 16 + l16) * Kdim + k + half * 8;
            FragB bfr;
            bfr.q[0] = *(const uint4*)(bp);
            bfr.q[1] = *(const uint4*)(bp + 16);
            acc[0][nc] = __builtin_amdgcn_wmma_f32_16x16x32_bf16(
                false, a0.v, false, bfr.v, (short)0, acc[0][nc], false, false);
            acc[1][nc] = __builtin_amdgcn_wmma_f32_16x16x32_bf16(
                false, a1.v, false, bfr.v, (short)0, acc[1][nc], false, false);
        }
    }

    // Epilogue. C-fragment: VGPR r -> row base + r + 8*half, lane l16 -> column.
    #pragma unroll
    for (int mt = 0; mt < 2; mt++) {
        const int mbase = m0 + mt * 16;
        #pragma unroll
        for (int nc = 0; nc < 4; nc++) {
            const int n  = n0 + nc * 16 + l16;
            const float bv = bias[n];
            if (OUT_MODE == 0) {
                unsigned short* out = (unsigned short*)Cout;
                #pragma unroll
                for (int r = 0; r < 8; r++) {
                    int row = mbase + half * 8 + r;
                    out[(size_t)row * Ndim + n] = f2bf(acc[mt][nc][r] + bv);
                }
            } else if (OUT_MODE == 2) {
                float* out = (float*)Cout;
                #pragma unroll
                for (int r = 0; r < 8; r++) {
                    int row = mbase + half * 8 + r;
                    out[(size_t)row * Ndim + n] = acc[mt][nc][r] + bv;
                }
            } else { // V transposed: 8 consecutive tokens -> one b128 store
                unsigned short* out = (unsigned short*)Cout;
                const int bidx = mbase / S_;
                const int s0   = (mbase % S_) + half * 8;
                const int h    = n / HD_;
                const int hd   = n % HD_;
                unsigned int w[4];
                #pragma unroll
                for (int i = 0; i < 4; i++) {
                    unsigned int lo = f2bf(acc[mt][nc][2 * i]     + bv);
                    unsigned int hi = f2bf(acc[mt][nc][2 * i + 1] + bv);
                    w[i] = lo | (hi << 16);
                }
                uint4 pk; pk.x = w[0]; pk.y = w[1]; pk.z = w[2]; pk.w = w[3];
                *(uint4*)(out + ((size_t)((bidx * H_ + h) * HD_ + hd)) * S_ + s0) = pk;
            }
        }
    }
}

// ---------------------------------------------------------------------------
// Flash attention: one wave per (b, h, 16-query tile), streaming 32-key tiles.
// Q,K in bf16 natural [B*S][D]; V transposed bf16 [B][H][HD][S].
// ---------------------------------------------------------------------------
__global__ __launch_bounds__(256) void attn_flash(
    const unsigned short* __restrict__ Qb, const unsigned short* __restrict__ Kb,
    const unsigned short* __restrict__ VT, const unsigned char* __restrict__ mask,
    unsigned short* __restrict__ CTX)
{
    __shared__ unsigned short pbuf[8 * 16 * 40]; // per-wave 16x32 P tile, pitch 40

    const int lane = threadIdx.x & 31;
    const int wave = threadIdx.x >> 5;
    const int half = lane >> 4;
    const int l16  = lane & 15;

    const int gw = blockIdx.x * 8 + wave;
    const int QT = S_ / 16;                 // 128 query tiles
    const int qt = gw % QT;
    const int h  = (gw / QT) % H_;
    const int b  = gw / (QT * H_);

    unsigned short* pl = &pbuf[wave * 16 * 40];

    // Q A-fragments for feature chunks [0,32) and [32,64)
    FragB qa[2];
    {
        const unsigned short* qrow =
            Qb + ((size_t)(b * S_ + qt * 16 + l16)) * D_ + h * HD_;
        #pragma unroll
        for (int c = 0; c < 2; c++) {
            qa[c].q[0] = *(const uint4*)(qrow + c * 32 + half * 8);
            qa[c].q[1] = *(const uint4*)(qrow + c * 32 + 16 + half * 8);
        }
    }

    v8f o[4];
    o[0] = {}; o[1] = {}; o[2] = {}; o[3] = {};
    float mrun[8], lrun[8];
    #pragma unroll
    for (int r = 0; r < 8; r++) { mrun[r] = -INFINITY; lrun[r] = 0.f; }

    const size_t mrowbase = ((size_t)b * S_ + (size_t)(qt * 16)) * S_;
    const unsigned short* vbase = VT + ((size_t)(b * H_ + h)) * HD_ * S_;

    for (int kt = 0; kt < S_; kt += 32) {
        // --- scores: S[16x32] = Q[16x64] @ K^T, as two 16x16 C-fragments ---
        v8f s[2];
        #pragma unroll
        for (int j = 0; j < 2; j++) {
            const unsigned short* krow =
                Kb + ((size_t)(b * S_ + kt + j * 16 + l16)) * D_ + h * HD_;
            FragB kb0, kb1;
            kb0.q[0] = *(const uint4*)(krow + half * 8);
            kb0.q[1] = *(const uint4*)(krow + 16 + half * 8);
            kb1.q[0] = *(const uint4*)(krow + 32 + half * 8);
            kb1.q[1] = *(const uint4*)(krow + 48 + half * 8);
            v8f z = {};
            z = __builtin_amdgcn_wmma_f32_16x16x32_bf16(
                false, qa[0].v, false, kb0.v, (short)0, z, false, false);
            z = __builtin_amdgcn_wmma_f32_16x16x32_bf16(
                false, qa[1].v, false, kb1.v, (short)0, z, false, false);
            s[j] = z;
        }

        // --- mask + scale + online softmax (row = r + 8*half, col = lane) ---
        #pragma unroll
        for (int r = 0; r < 8; r++) {
            const int qrow = r + half * 8;
            const unsigned char* mrow = mask + mrowbase + (size_t)qrow * S_ + kt;
            float s0 = mrow[l16]      ? -1e15f : s[0][r] * 0.125f;
            float s1 = mrow[16 + l16] ? -1e15f : s[1][r] * 0.125f;

            float t = fmaxf(s0, s1);
            t = fmaxf(t, __shfl_xor(t, 1, 32));
            t = fmaxf(t, __shfl_xor(t, 2, 32));
            t = fmaxf(t, __shfl_xor(t, 4, 32));
            t = fmaxf(t, __shfl_xor(t, 8, 32));

            const float nm    = fmaxf(mrun[r], t);
            const float alpha = __expf(mrun[r] - nm);
            const float p0 = __expf(s0 - nm);
            const float p1 = __expf(s1 - nm);
            float rs = p0 + p1;
            rs += __shfl_xor(rs, 1, 32);
            rs += __shfl_xor(rs, 2, 32);
            rs += __shfl_xor(rs, 4, 32);
            rs += __shfl_xor(rs, 8, 32);

            lrun[r] = lrun[r] * alpha + rs;
            mrun[r] = nm;
            #pragma unroll
            for (int nc = 0; nc < 4; nc++) o[nc][r] *= alpha;

            pl[qrow * 40 + l16]      = f2bf(p0);
            pl[qrow * 40 + 16 + l16] = f2bf(p1);
        }

        // --- transpose P C-layout -> A-layout through per-wave LDS tile ---
        FragB pa;
        __builtin_memcpy(&pa.q[0], pl + l16 * 40 + half * 8,      16);
        __builtin_memcpy(&pa.q[1], pl + l16 * 40 + 16 + half * 8, 16);

        // --- ctx += P[16x32] @ V[32x64], V^T rows give contiguous B-frags ---
        #pragma unroll
        for (int nc = 0; nc < 4; nc++) {
            const unsigned short* vrow = vbase + (size_t)(nc * 16 + l16) * S_ + kt;
            FragB vb;
            vb.q[0] = *(const uint4*)(vrow + half * 8);
            vb.q[1] = *(const uint4*)(vrow + 16 + half * 8);
            o[nc] = __builtin_amdgcn_wmma_f32_16x16x32_bf16(
                false, pa.v, false, vb.v, (short)0, o[nc], false, false);
        }
    }

    // --- normalize and store ctx (bf16, natural [B*S][D]) ---
    #pragma unroll
    for (int r = 0; r < 8; r++) {
        const float inv = 1.0f / lrun[r];
        const int row = b * S_ + qt * 16 + r + half * 8;
        #pragma unroll
        for (int nc = 0; nc < 4; nc++) {
            CTX[(size_t)row * D_ + h * HD_ + nc * 16 + l16] = f2bf(o[nc][r] * inv);
        }
    }
}

// ---------------------------------------------------------------------------
extern "C" void kernel_launch(void* const* d_in, const int* in_sizes, int n_in,
                              void* d_out, int out_size, void* d_ws, size_t ws_size,
                              hipStream_t stream) {
    (void)in_sizes; (void)n_in; (void)out_size; (void)ws_size;

    const float*         queries = (const float*)d_in[0];
    const float*         keys    = (const float*)d_in[1];
    const float*         values  = (const float*)d_in[2];
    const unsigned char* mask    = (const unsigned char*)d_in[3];
    const float* Wq = (const float*)d_in[4];
    const float* bq = (const float*)d_in[5];
    const float* Wk = (const float*)d_in[6];
    const float* bk = (const float*)d_in[7];
    const float* Wv = (const float*)d_in[8];
    const float* bv = (const float*)d_in[9];
    const float* Wo = (const float*)d_in[10];
    const float* bo = (const float*)d_in[11];

    // Workspace (bf16): 4 transposed weights + Xq/Xk/Xv + Q + K + V^T.
    // CTX aliases Xq (dead after the Q projection GEMM).
    unsigned short* WqT = (unsigned short*)d_ws;
    unsigned short* WkT = WqT + (size_t)D_ * D_;
    unsigned short* WvT = WkT + (size_t)D_ * D_;
    unsigned short* WoT = WvT + (size_t)D_ * D_;
    unsigned short* Xq  = WoT + (size_t)D_ * D_;
    unsigned short* Xk  = Xq  + (size_t)MTOT * D_;
    unsigned short* Xv  = Xk  + (size_t)MTOT * D_;
    unsigned short* Qb  = Xv  + (size_t)MTOT * D_;
    unsigned short* Kb  = Qb  + (size_t)MTOT * D_;
    unsigned short* Vt  = Kb  + (size_t)MTOT * D_;
    unsigned short* CTX = Xq;  // alias

    const dim3 tb(256);

    const int wtElems = D_ * D_;
    const int wtGrid  = (wtElems + 255) / 256;
    wt_transpose_bf16<<<wtGrid, tb, 0, stream>>>(Wq, WqT, D_, D_);
    wt_transpose_bf16<<<wtGrid, tb, 0, stream>>>(Wk, WkT, D_, D_);
    wt_transpose_bf16<<<wtGrid, tb, 0, stream>>>(Wv, WvT, D_, D_);
    wt_transpose_bf16<<<wtGrid, tb, 0, stream>>>(Wo, WoT, D_, D_);

    const int n8 = (MTOT * D_) / 8;        // 1M threads per tensor
    const int cvtGrid = (n8 + 255) / 256;
    cvt_f32_bf16_vec8<<<cvtGrid, tb, 0, stream>>>(queries, Xq, n8);
    cvt_f32_bf16_vec8<<<cvtGrid, tb, 0, stream>>>(keys,    Xk, n8);
    cvt_f32_bf16_vec8<<<cvtGrid, tb, 0, stream>>>(values,  Xv, n8);

    const dim3 ggrid(D_ / 128, MTOT / 128);   // (8, 64)
    gemm32x64<0><<<ggrid, tb, 0, stream>>>(Xq, WqT, bq, Qb, MTOT, D_, D_);
    gemm32x64<0><<<ggrid, tb, 0, stream>>>(Xk, WkT, bk, Kb, MTOT, D_, D_);
    gemm32x64<1><<<ggrid, tb, 0, stream>>>(Xv, WvT, bv, Vt, MTOT, D_, D_);

    attn_flash<<<(B_ * H_ * (S_ / 16)) / 8, tb, 0, stream>>>(Qb, Kb, Vt, mask, CTX);

    gemm32x64<2><<<ggrid, tb, 0, stream>>>(CTX, WoT, bo, d_out, MTOT, D_, D_);
}
